// GGRU_v0_68599217652304
// MI455X (gfx1250) — matrix operand order repaired and checked by reading, not compiled, not run on hardware
//
#include <hip/hip_runtime.h>
#include <hip/hip_bf16.h>

// ---------------------------------------------------------------------------
// Problem constants (from reference): B=256, T=512, D=128, Hd=256
// ---------------------------------------------------------------------------
#define BB 256
#define TT 512
#define DD 128
#define HD 256
#define KT_X 4   // k-tiles (of 32) for D=128
#define KT_H 8   // k-tiles (of 32) for Hd=256
#define NT_H 16  // n-tiles (of 16) for Hd=256

typedef __attribute__((ext_vector_type(16))) __bf16 v16bf;
typedef __attribute__((ext_vector_type(8)))  float v8f;

static __device__ __forceinline__ unsigned short f2bf(float f) {
  // round-to-nearest-even fp32 -> bf16 (used only in the one-shot pack kernel)
  unsigned int u = __float_as_uint(f);
  u += 0x7FFFu + ((u >> 16) & 1u);
  return (unsigned short)(u >> 16);
}
// fast sigmoid / tanh on native v_exp_f32 + v_rcp_f32
static __device__ __forceinline__ float sigm(float x) {
  return __builtin_amdgcn_rcpf(1.0f + __expf(-x));
}
static __device__ __forceinline__ float ftanh(float x) {
  return 1.0f - 2.0f * __builtin_amdgcn_rcpf(1.0f + __expf(2.0f * x));
}
static __device__ __forceinline__ v8f zero8() {
  v8f v;
#pragma unroll
  for (int i = 0; i < 8; ++i) v[i] = 0.0f;
  return v;
}
static __device__ __forceinline__ v8f wmma_bf16(v16bf a, v16bf b, v8f c) {
  return __builtin_amdgcn_wmma_f32_16x16x32_bf16(false, a, false, b,
                                                 (short)0, c, false, false);
}

// A-operand (16x32 bf16) fetch from packed LDS: lane-contiguous 32B.
static __device__ __forceinline__ v16bf loadA(const __bf16* buf, int kt,
                                              int lane) {
  return *reinterpret_cast<const v16bf*>(buf + ((kt * 32 + lane) * 16));
}
// B-operand (32x16 bf16) fetch from packed global weights: lane-contiguous 32B.
static __device__ __forceinline__ v16bf loadB(const unsigned short* wm,
                                              int kt, int nt, int lane) {
  return *reinterpret_cast<const v16bf*>(wm + (((kt * NT_H + nt) * 32 + lane) * 16));
}
// B-operand fetch from the LDS weight cache (tile-contiguous layout).
static __device__ __forceinline__ v16bf loadBL(const __bf16* buf, int tile,
                                               int lane) {
  return *reinterpret_cast<const v16bf*>(buf + ((tile * 32 + lane) * 16));
}
// Scatter one C-layout v8f (M=r+(lane>=16?8:0), N=colBase+lane%16) into the
// A-operand packed LDS layout (value becomes a K-column of the next matmul).
static __device__ __forceinline__ void packA(__bf16* buf, v8f v, int colBase,
                                             int lane) {
  const int Mbase = (lane >= 16) ? 8 : 0;
  const int K  = colBase + (lane & 15);
  const int kt = K >> 5;
  const int kk = K & 31;
  const int h  = (kk & 7) + ((kk >= 16) ? 8 : 0);
  const int la = ((kk & 15) >= 8) ? 16 : 0;
#pragma unroll
  for (int r = 0; r < 8; ++r) {
    buf[((kt * 32 + (la + Mbase + r)) * 16) + h] = (__bf16)v[r];
  }
}
// Store 8 consecutive-K X values (two float4) through precomputed A-layout idx.
static __device__ __forceinline__ void packX8(__bf16* buf, const int* sidx,
                                              float4 a, float4 b) {
  buf[sidx[0]] = (__bf16)a.x; buf[sidx[1]] = (__bf16)a.y;
  buf[sidx[2]] = (__bf16)a.z; buf[sidx[3]] = (__bf16)a.w;
  buf[sidx[4]] = (__bf16)b.x; buf[sidx[5]] = (__bf16)b.y;
  buf[sidx[6]] = (__bf16)b.z; buf[sidx[7]] = (__bf16)b.w;
}

// ---------------------------------------------------------------------------
// Kernel 0: pack one fp32 [Krows x 256] weight (row-major) into bf16 WMMA
// B-operand tiles: dst[((kt*16+nt)*32+lane)*16+h] =
//   W[kt*32+(h&7)+(lane>=16?8:0)+(h>=8?16:0)][nt*16 + lane%16]
// ---------------------------------------------------------------------------
__global__ void ggru_pack_w(const float* __restrict__ src,
                            unsigned short* __restrict__ dst, int Krows) {
  int e = blockIdx.x * 256 + threadIdx.x;
  if (e >= Krows * 256) return;
  int h    = e & 15;
  int lane = (e >> 4) & 31;
  int tn   = e >> 9;
  int nt   = tn & 15;
  int kt   = tn >> 4;
  int K = kt * 32 + (h & 7) + ((lane >= 16) ? 8 : 0) + ((h >= 8) ? 16 : 0);
  int n = nt * 16 + (lane & 15);
  dst[e] = f2bf(src[K * 256 + n]);
}

// ---------------------------------------------------------------------------
// Kernel 1: sequential dual-GRU scan. grid = 16 WGs x 256 threads; each WG
// owns 16 batch rows; each of its 8 waves owns 32 hidden columns.
// Weight tiering: fus_W_t in VGPRs, W_hh_* in a 256KB LDS cache, the rest
// streamed from L2 each step via an opaque (non-hoistable) pointer.
// Accumulators start at zero; biases are folded into the activations.
// ---------------------------------------------------------------------------
__global__ __launch_bounds__(256, 1) void ggru_scan(
    const float* __restrict__ tt, const float* __restrict__ mk,
    const unsigned short* __restrict__ wpk,
    const float* __restrict__ bzt, const float* __restrict__ brt,
    const float* __restrict__ bht, const float* __restrict__ bzm,
    const float* __restrict__ brm, const float* __restrict__ bhm,
    const float* __restrict__ fbias, float* __restrict__ out) {
  __shared__ __align__(32) __bf16 sXt[2][KT_X * 32 * 16];
  __shared__ __align__(32) __bf16 sXm[2][KT_X * 32 * 16];
  __shared__ __align__(32) __bf16 sHp[KT_H * 32 * 16];
  __shared__ __align__(32) __bf16 sRt[KT_H * 32 * 16];
  __shared__ __align__(32) __bf16 sRm[KT_H * 32 * 16];
  __shared__ __align__(32) __bf16 sHt[KT_H * 32 * 16];
  __shared__ __align__(32) __bf16 sHm[KT_H * 32 * 16];
  // LDS weight cache: per wave, W_hh_t and W_hh_m tiles for its 2 n-tiles:
  // [wave][mat(2)][nti(2)][kt(8)] tiles of 32x16 bf16 = 256 KB total
  __shared__ __align__(32) __bf16 sWhh[8 * 2 * 2 * 8 * 32 * 16];

  const int tid  = threadIdx.x;
  const int lane = tid & 31;
  const int w    = tid >> 5;
  const int b0   = blockIdx.x * 16;

  // packed weight offsets (ushorts): 6 x-weights then 8 h-weights
  const int oWxzT = 0 * 32768, oWxrT = 1 * 32768, oWxhT = 2 * 32768;
  const int oWxzM = 3 * 32768, oWxrM = 4 * 32768, oWxhM = 5 * 32768;
  const int hb = 6 * 32768;
  const int oWhzT = hb + 0 * 65536, oWhrT = hb + 1 * 65536;
  const int oWhhT = hb + 2 * 65536, oWhzM = hb + 3 * 65536;
  const int oWhrM = hb + 4 * 65536, oWhhM = hb + 5 * 65536;
  const int oFWt  = hb + 6 * 65536, oFWm  = hb + 7 * 65536;

  const int nt0 = w * 2, nt1 = w * 2 + 1;
  const int col0 = nt0 * 16 + (lane & 15);
  const int col1 = nt1 * 16 + (lane & 15);
  const int mrow = (lane >= 16) ? 8 : 0;

  // per-column scalar biases (folded into activations, not accumulators)
  const float bzt0 = bzt[col0], bzt1 = bzt[col1];
  const float brt0 = brt[col0], brt1 = brt[col1];
  const float bht0 = bht[col0], bht1 = bht[col1];
  const float bzm0 = bzm[col0], bzm1 = bzm[col1];
  const float brm0 = brm[col0], brm1 = brm[col1];
  const float bhm0 = bhm[col0], bhm1 = bhm[col1];
  const float fb0  = fbias[col0], fb1 = fbias[col1];

  // ---- VGPR weight cache: fus_W_t tiles (16 tiles = 128 VGPRs) ----
  v16bf rFt[2][KT_H];
#pragma unroll
  for (int kt = 0; kt < KT_H; ++kt) {
    rFt[0][kt] = loadB(wpk + oFWt, kt, nt0, lane);
    rFt[1][kt] = loadB(wpk + oFWt, kt, nt1, lane);
  }
  // ---- LDS weight cache: W_hh_t / W_hh_m tiles for this wave ----
  {
    __bf16* dst = sWhh + (size_t)w * (2 * 2 * KT_H * 32 * 16);
#pragma unroll
    for (int m2 = 0; m2 < 2; ++m2) {
      const unsigned short* src = wpk + (m2 ? oWhhM : oWhhT);
#pragma unroll
      for (int nti = 0; nti < 2; ++nti) {
#pragma unroll
        for (int kt = 0; kt < KT_H; ++kt) {
          v16bf v = loadB(src, kt, nti ? nt1 : nt0, lane);
          int tile = (m2 * 2 + nti) * KT_H + kt;
          *reinterpret_cast<v16bf*>(dst + ((tile * 32 + lane) * 16)) = v;
        }
      }
    }
  }
  const __bf16* sWhhT0 = sWhh + (size_t)w * (2 * 2 * KT_H * 32 * 16);
  const __bf16* sWhhT1 = sWhhT0 + 1 * KT_H * 32 * 16;
  const __bf16* sWhhM0 = sWhhT0 + 2 * KT_H * 32 * 16;
  const __bf16* sWhhM1 = sWhhT0 + 3 * KT_H * 32 * 16;

  // per-thread X slab addressing: 8 consecutive K of one row M
  const int xbase = tid * 8;       // 2048 elems = 256 threads * 8
  const int xM    = xbase >> 7;    // row 0..15
  const int xK0   = xbase & 127;
  const float* ptBase = tt + (size_t)(b0 + xM) * TT * DD + xK0;
  const float* pmBase = mk + (size_t)(b0 + xM) * TT * DD + xK0;
  int sidx[8];
#pragma unroll
  for (int j = 0; j < 8; ++j) {
    int K  = xK0 + j;
    int kt = K >> 5, kk = K & 31;
    int h  = (kk & 7) + ((kk >= 16) ? 8 : 0);
    int la = xM + (((kk & 15) >= 8) ? 16 : 0);
    sidx[j] = (kt * 32 + la) * 16 + h;
  }

  // H0 = 0
  for (int i = tid; i < KT_H * 32 * 16; i += 256) sHp[i] = (__bf16)0.0f;
  v8f hp0 = zero8(), hp1 = zero8();

  // prologue: load X(0) into registers
  float4 rxt0 = *(const float4*)(ptBase);
  float4 rxt1 = *(const float4*)(ptBase + 4);
  float4 rxm0 = *(const float4*)(pmBase);
  float4 rxm1 = *(const float4*)(pmBase + 4);

  // opaque weight base: compiler cannot hoist streamed loads out of the loop
  const unsigned short* wp = wpk;

  for (int t = 0; t < TT; ++t) {
    const int cur = t & 1;
    // ---- commit prefetched X(t) into LDS A-layout, then prefetch X(t+1) ----
    packX8(sXt[cur], sidx, rxt0, rxt1);
    packX8(sXm[cur], sidx, rxm0, rxm1);
    if (t + 1 < TT) {
      const float* pt = ptBase + (size_t)(t + 1) * DD;
      const float* pm = pmBase + (size_t)(t + 1) * DD;
      rxt0 = *(const float4*)(pt);
      rxt1 = *(const float4*)(pt + 4);
      rxm0 = *(const float4*)(pm);
      rxm1 = *(const float4*)(pm + 4);
    }
    // break loop-invariance of streamed weight loads (prevents hoist+spill)
    asm volatile("" : "+s"(wp));
    const unsigned short* cWxzT = wp + oWxzT;
    const unsigned short* cWxrT = wp + oWxrT;
    const unsigned short* cWxhT = wp + oWxhT;
    const unsigned short* cWxzM = wp + oWxzM;
    const unsigned short* cWxrM = wp + oWxrM;
    const unsigned short* cWxhM = wp + oWxhM;
    const unsigned short* cWhzT = wp + oWhzT;
    const unsigned short* cWhrT = wp + oWhrT;
    const unsigned short* cWhzM = wp + oWhzM;
    const unsigned short* cWhrM = wp + oWhrM;
    const unsigned short* cFWm  = wp + oFWm;
    __syncthreads();  // sX[cur] + previous step's sHp now visible to all waves

    // ---- Phase A: z and r gates for both GRUs (zero-init accumulators) ----
    v8f aZt0 = zero8(), aZt1 = zero8();
    v8f aRt0 = zero8(), aRt1 = zero8();
    v8f aZm0 = zero8(), aZm1 = zero8();
    v8f aRm0 = zero8(), aRm1 = zero8();
#pragma unroll
    for (int kt = 0; kt < KT_X; ++kt) {
      v16bf at = loadA(sXt[cur], kt, lane);
      v16bf am = loadA(sXm[cur], kt, lane);
      aZt0 = wmma_bf16(at, loadB(cWxzT, kt, nt0, lane), aZt0);
      aZt1 = wmma_bf16(at, loadB(cWxzT, kt, nt1, lane), aZt1);
      aRt0 = wmma_bf16(at, loadB(cWxrT, kt, nt0, lane), aRt0);
      aRt1 = wmma_bf16(at, loadB(cWxrT, kt, nt1, lane), aRt1);
      aZm0 = wmma_bf16(am, loadB(cWxzM, kt, nt0, lane), aZm0);
      aZm1 = wmma_bf16(am, loadB(cWxzM, kt, nt1, lane), aZm1);
      aRm0 = wmma_bf16(am, loadB(cWxrM, kt, nt0, lane), aRm0);
      aRm1 = wmma_bf16(am, loadB(cWxrM, kt, nt1, lane), aRm1);
    }
#pragma unroll
    for (int kt = 0; kt < KT_H; ++kt) {
      v16bf ah = loadA(sHp, kt, lane);
      aZt0 = wmma_bf16(ah, loadB(cWhzT, kt, nt0, lane), aZt0);
      aZt1 = wmma_bf16(ah, loadB(cWhzT, kt, nt1, lane), aZt1);
      aRt0 = wmma_bf16(ah, loadB(cWhrT, kt, nt0, lane), aRt0);
      aRt1 = wmma_bf16(ah, loadB(cWhrT, kt, nt1, lane), aRt1);
      aZm0 = wmma_bf16(ah, loadB(cWhzM, kt, nt0, lane), aZm0);
      aZm1 = wmma_bf16(ah, loadB(cWhzM, kt, nt1, lane), aZm1);
      aRm0 = wmma_bf16(ah, loadB(cWhrM, kt, nt0, lane), aRm0);
      aRm1 = wmma_bf16(ah, loadB(cWhrM, kt, nt1, lane), aRm1);
    }
    // sigmoids (bias folded in); R*H packed for the candidate matmuls
    v8f rh;
#pragma unroll
    for (int i = 0; i < 8; ++i) rh[i] = sigm(aRt0[i] + brt0) * hp0[i];
    packA(sRt, rh, nt0 * 16, lane);
#pragma unroll
    for (int i = 0; i < 8; ++i) rh[i] = sigm(aRt1[i] + brt1) * hp1[i];
    packA(sRt, rh, nt1 * 16, lane);
#pragma unroll
    for (int i = 0; i < 8; ++i) rh[i] = sigm(aRm0[i] + brm0) * hp0[i];
    packA(sRm, rh, nt0 * 16, lane);
#pragma unroll
    for (int i = 0; i < 8; ++i) rh[i] = sigm(aRm1[i] + brm1) * hp1[i];
    packA(sRm, rh, nt1 * 16, lane);
    v8f zt0, zt1, zm0, zm1;
#pragma unroll
    for (int i = 0; i < 8; ++i) {
      zt0[i] = sigm(aZt0[i] + bzt0); zt1[i] = sigm(aZt1[i] + bzt1);
      zm0[i] = sigm(aZm0[i] + bzm0); zm1[i] = sigm(aZm1[i] + bzm1);
    }
    __syncthreads();

    // ---- Phase B: candidate states, per-GRU hidden updates ----
    v8f aHt0 = zero8(), aHt1 = zero8();
    v8f aHm0 = zero8(), aHm1 = zero8();
#pragma unroll
    for (int kt = 0; kt < KT_X; ++kt) {
      v16bf at = loadA(sXt[cur], kt, lane);
      v16bf am = loadA(sXm[cur], kt, lane);
      aHt0 = wmma_bf16(at, loadB(cWxhT, kt, nt0, lane), aHt0);
      aHt1 = wmma_bf16(at, loadB(cWxhT, kt, nt1, lane), aHt1);
      aHm0 = wmma_bf16(am, loadB(cWxhM, kt, nt0, lane), aHm0);
      aHm1 = wmma_bf16(am, loadB(cWxhM, kt, nt1, lane), aHm1);
    }
#pragma unroll
    for (int kt = 0; kt < KT_H; ++kt) {
      v16bf art = loadA(sRt, kt, lane);
      v16bf arm = loadA(sRm, kt, lane);
      aHt0 = wmma_bf16(art, loadBL(sWhhT0, kt, lane), aHt0);
      aHt1 = wmma_bf16(art, loadBL(sWhhT1, kt, lane), aHt1);
      aHm0 = wmma_bf16(arm, loadBL(sWhhM0, kt, lane), aHm0);
      aHm1 = wmma_bf16(arm, loadBL(sWhhM1, kt, lane), aHm1);
    }
    v8f Ht0, Ht1, Hm0, Hm1;
#pragma unroll
    for (int i = 0; i < 8; ++i) {
      Ht0[i] = zt0[i] * hp0[i] + (1.0f - zt0[i]) * ftanh(aHt0[i] + bht0);
      Ht1[i] = zt1[i] * hp1[i] + (1.0f - zt1[i]) * ftanh(aHt1[i] + bht1);
      Hm0[i] = zm0[i] * hp0[i] + (1.0f - zm0[i]) * ftanh(aHm0[i] + bhm0);
      Hm1[i] = zm1[i] * hp1[i] + (1.0f - zm1[i]) * ftanh(aHm1[i] + bhm1);
    }
    packA(sHt, Ht0, nt0 * 16, lane);
    packA(sHt, Ht1, nt1 * 16, lane);
    packA(sHm, Hm0, nt0 * 16, lane);
    packA(sHm, Hm1, nt1 * 16, lane);
    __syncthreads();

    // ---- Phase C: fusion gate and new hidden state ----
    v8f aF0 = zero8(), aF1 = zero8();
#pragma unroll
    for (int kt = 0; kt < KT_H; ++kt) {
      v16bf aht = loadA(sHt, kt, lane);
      v16bf ahm = loadA(sHm, kt, lane);
      aF0 = wmma_bf16(aht, rFt[0][kt], aF0);
      aF1 = wmma_bf16(aht, rFt[1][kt], aF1);
      aF0 = wmma_bf16(ahm, loadB(cFWm, kt, nt0, lane), aF0);
      aF1 = wmma_bf16(ahm, loadB(cFWm, kt, nt1, lane), aF1);
    }
    v8f Hn0, Hn1;
#pragma unroll
    for (int i = 0; i < 8; ++i) {
      float z0 = sigm(aF0[i] + fb0);
      float z1 = sigm(aF1[i] + fb1);
      Hn0[i] = z0 * Ht0[i] + (1.0f - z0) * Hm0[i];
      Hn1[i] = z1 * Ht1[i] + (1.0f - z1) * Hm1[i];
    }
    // stream outputs[b0+m, t, col] (write-once: non-temporal)
#pragma unroll
    for (int r = 0; r < 8; ++r) {
      const int m = mrow + r;
      float* o = out + ((size_t)(b0 + m) * TT + t) * HD;
      __builtin_nontemporal_store(Hn0[r], o + col0);
      __builtin_nontemporal_store(Hn1[r], o + col1);
    }
    // new H becomes next step's matmul operand + register-resident state;
    // safe: all waves finished reading old sHp before the post-Phase-A barrier
    packA(sHp, Hn0, nt0 * 16, lane);
    packA(sHp, Hn1, nt1 * 16, lane);
    hp0 = Hn0;
    hp1 = Hn1;
  }

  // H_final = outputs[:, T-1, :] appended after the [B,T,HD] block
  float* hf = out + (size_t)BB * TT * HD;
#pragma unroll
  for (int r = 0; r < 8; ++r) {
    const int m = mrow + r;
    hf[(size_t)(b0 + m) * HD + col0] = hp0[r];
    hf[(size_t)(b0 + m) * HD + col1] = hp1[r];
  }
}

// ---------------------------------------------------------------------------
extern "C" void kernel_launch(void* const* d_in, const int* in_sizes, int n_in,
                              void* d_out, int out_size, void* d_ws,
                              size_t ws_size, hipStream_t stream) {
  (void)in_sizes; (void)n_in; (void)out_size; (void)ws_size;
  const float* tt = (const float*)d_in[0];
  const float* mk = (const float*)d_in[1];
  unsigned short* wpk = (unsigned short*)d_ws;

  // pack the 14 weight matrices into bf16 B-operand tiles in workspace
  const int srcIdx[14] = {2, 5, 8, 11, 14, 17, 3, 6, 9, 12, 15, 18, 20, 21};
  const int krows[14]  = {128, 128, 128, 128, 128, 128,
                          256, 256, 256, 256, 256, 256, 256, 256};
  size_t off = 0;
  for (int i = 0; i < 14; ++i) {
    int elems = krows[i] * 256;
    ggru_pack_w<<<(elems + 255) / 256, 256, 0, stream>>>(
        (const float*)d_in[srcIdx[i]], wpk + off, krows[i]);
    off += (size_t)elems;
  }

  ggru_scan<<<BB / 16, 256, 0, stream>>>(
      tt, mk, wpk,
      (const float*)d_in[4], (const float*)d_in[7], (const float*)d_in[10],
      (const float*)d_in[13], (const float*)d_in[16], (const float*)d_in[19],
      (const float*)d_in[22], (float*)d_out);
}
